// SequencePackingAttention_77988016161208
// MI455X (gfx1250) — matrix operand order repaired and checked by reading, not compiled, not run on hardware
//
#include <hip/hip_runtime.h>
#include <hip/hip_bf16.h>

// ---------------------------------------------------------------------------
// Packed-sequence MHA for MI455X (gfx1250, wave32, WMMA).
// Compute-bound (~25 GFLOP vs ~30 MB traffic) -> all matmuls via
// v_wmma_f32_16x16x32_bf16 (fp32 accumulate). Block-diagonal bias handled
// structurally (segments 512/640/384/256/256, all 32-aligned -> no masking).
// GEMM uses double-buffered LDS B-panels staged with the CDNA5 async path
// (global_load_async_to_lds_b128 + s_wait_asynccnt), bypassing VGPRs.
// ---------------------------------------------------------------------------

typedef __attribute__((ext_vector_type(16))) __bf16 v16bf;
typedef __attribute__((ext_vector_type(8)))  float  v8f;
typedef __attribute__((ext_vector_type(4)))  float  v4f;

#define N_TOK 2048
#define CDIM  1024
#define HEADS 16
#define DH    64

union ABuf {             // one WMMA 16-bit operand: 16 halves/lane = 32 bytes
  v16bf v;
  v4f   f4[2];
};

__device__ __forceinline__ unsigned short f2bf(float f) {
  unsigned int u = __float_as_uint(f);
  u += 0x7FFFu + ((u >> 16) & 1u);          // round-to-nearest-even
  return (unsigned short)(u >> 16);
}

__device__ __forceinline__ v8f wmma_bf16(v16bf a, v16bf b, v8f c) {
  // (neg_a, A, neg_b, B, c_mod, C, reuse_a, reuse_b)
  return __builtin_amdgcn_wmma_f32_16x16x32_bf16(false, a, false, b,
                                                 (short)0, c, false, false);
}

// CDNA5 async copy: 16 bytes/lane, global -> LDS, tracked by ASYNCcnt.
// Generic pointers to __shared__ have the LDS offset in addr[31:0] (ISA 10.2).
__device__ __forceinline__ void async_g2l_b128(void* lds, const void* gptr) {
  unsigned lds_off = (unsigned)(unsigned long long)lds;
  unsigned long long g = (unsigned long long)gptr;
  asm volatile("global_load_async_to_lds_b128 %0, %1, off"
               :: "v"(lds_off), "v"(g) : "memory");
}

// ---------------------------------------------------------------------------
// fp32 -> bf16 conversion (vectorized x4)
// ---------------------------------------------------------------------------
__global__ void cvt_f32_bf16(const float* __restrict__ in,
                             unsigned short* __restrict__ out, int n4) {
  int i = blockIdx.x * blockDim.x + threadIdx.x;
  if (i >= n4) return;
  v4f f = *(const v4f*)(in + (size_t)i * 4);
  uint2 o;
  o.x = (unsigned)f2bf(f.x) | ((unsigned)f2bf(f.y) << 16);
  o.y = (unsigned)f2bf(f.z) | ((unsigned)f2bf(f.w) << 16);
  *(uint2*)(out + (size_t)i * 4) = o;
}

// ---------------------------------------------------------------------------
// bf16 GEMM: C[M x Ncol] = A[M x K] * B[K x Ncol], fp32 accum via WMMA.
// Workgroup: 8 waves, 256x64 tile; wave w -> rows [32w, 32w+32) x 64 cols
// (two A operands, 8 accumulators -> 8 WMMAs per 32-k step, B reused x2).
// B k-panels (32x64 bf16 = 4KB) double-buffered in LDS, staged by async
// global->LDS copies overlapped with WMMA compute on the other panel.
// mode 0: QKV epilogue -> scatter Q (scaled), K^T, V head-major bf16.
// mode 1: proj epilogue -> fp32 out + bias.
// ---------------------------------------------------------------------------
__global__ void __launch_bounds__(256)
gemm_bf16(const unsigned short* __restrict__ A,
          const unsigned short* __restrict__ B,
          int K, int Ncol, int mode,
          unsigned short* __restrict__ qOut,     // [H][N][DH]
          unsigned short* __restrict__ kTOut,    // [H][DH][N]
          unsigned short* __restrict__ vOut,     // [H][N][DH]
          float* __restrict__ projOut,           // [M][Ncol]
          const float* __restrict__ bias) {
  __shared__ __align__(16) unsigned short Bsh[2][32 * 64];

  const int lane = threadIdx.x & 31;
  const int wave = threadIdx.x >> 5;
  const int lm   = lane & 15;
  const int half = lane >> 4;
  const int kb   = half * 8;                 // A-operand K base per lane half
  const int row0 = blockIdx.y * 256 + wave * 32;
  const int col0 = blockIdx.x * 64;

  // LDS staging coords: 2048 bf16 / 256 threads = 8 bf16 (16B) per thread
  const int e  = threadIdx.x * 8;
  const int sr = e >> 6;                     // k offset 0..31
  const int sc = e & 63;                     // col offset
  const unsigned short* bsrc = B + (size_t)sr * Ncol + col0 + sc;

  v8f acc0[4] = {}, acc1[4] = {};            // rows rg=0 / rg=1, 4 col-chunks

  const int nsteps = K / 32;
  // Prologue: async-stage panel 0 (k=0)
  async_g2l_b128(&Bsh[0][e], bsrc);

  for (int i = 0; i < nsteps; ++i) {
    const int k0  = i * 32;
    const int buf = i & 1;
    if (i + 1 < nsteps) {
      // Overlap: stage next panel while computing this one.
      async_g2l_b128(&Bsh[buf ^ 1][e], bsrc + (size_t)(k0 + 32) * Ncol);
      asm volatile("s_wait_asynccnt 0x1" ::: "memory");  // own panel i done
    } else {
      asm volatile("s_wait_asynccnt 0x0" ::: "memory");
    }
    __syncthreads();   // every wave's chunk of panel i landed in LDS

    // A operands: lane = row (lane%16), halves = K {kb..kb+7, kb+16..kb+23}
    ABuf a0, a1;
    const unsigned short* ar0 = A + (size_t)(row0 + lm) * K + k0 + kb;
    const unsigned short* ar1 = A + (size_t)(row0 + 16 + lm) * K + k0 + kb;
    a0.f4[0] = *(const v4f*)(ar0); a0.f4[1] = *(const v4f*)(ar0 + 16);
    a1.f4[0] = *(const v4f*)(ar1); a1.f4[1] = *(const v4f*)(ar1 + 16);
    if (k0 + 32 < K) {  // gfx1250 global_prefetch for next A chunks
      __builtin_prefetch(ar0 + 32, 0, 1);
      __builtin_prefetch(ar1 + 32, 0, 1);
    }

    // B operand: lane = column (lane%16), halves = K {kb..kb+7, kb+16..kb+23}
    const unsigned short* bls = &Bsh[buf][kb * 64 + lm];
#pragma unroll
    for (int nt = 0; nt < 4; ++nt) {
      ABuf b;
      for (int j = 0; j < 16; ++j)
        b.v[j] = __builtin_bit_cast(__bf16, bls[((j & 7) + (j >> 3) * 16) * 64 + nt * 16]);
      acc0[nt] = wmma_bf16(a0.v, b.v, acc0[nt]);
      acc1[nt] = wmma_bf16(a1.v, b.v, acc1[nt]);
    }
    __syncthreads();   // all reads of panel i done before it is re-staged
  }

  if (mode == 0) {
    // col0 is 64-aligned -> whole block belongs to one (t, h)
    const int t = col0 >> 10;                // 0=Q 1=K 2=V
    const int h = (col0 & 1023) >> 6;
#pragma unroll
    for (int rg = 0; rg < 2; ++rg) {
#pragma unroll
      for (int nt = 0; nt < 4; ++nt) {
        const int d = nt * 16 + lm;
#pragma unroll
        for (int r = 0; r < 8; ++r) {
          const float val = rg ? acc1[nt][r] : acc0[nt][r];
          const int grow = row0 + rg * 16 + r + half * 8;
          if (t == 0)
            qOut[((size_t)h * N_TOK + grow) * DH + d] = f2bf(val * 0.125f);
          else if (t == 1)
            kTOut[((size_t)h * DH + d) * N_TOK + grow] = f2bf(val);
          else
            vOut[((size_t)h * N_TOK + grow) * DH + d] = f2bf(val);
        }
      }
    }
  } else {
#pragma unroll
    for (int rg = 0; rg < 2; ++rg) {
#pragma unroll
      for (int nt = 0; nt < 4; ++nt) {
        const int c = col0 + nt * 16 + lm;
        const float bv = bias[c];
#pragma unroll
        for (int r = 0; r < 8; ++r) {
          const int grow = row0 + rg * 16 + r + half * 8;
          projOut[(size_t)grow * Ncol + c] = (rg ? acc1[nt][r] : acc0[nt][r]) + bv;
        }
      }
    }
  }
}

// ---------------------------------------------------------------------------
// Flash attention over packed segments. One wave per (head, 16-row tile);
// iterates 32-column K/V blocks within the tile's segment only.
// C/D tile layout: VGPR r <-> rows {r, r+8}, lanes = columns, so per-lane row
// stats (mi/li) reduce across 16 lanes via __shfl_xor(width=16).
// ---------------------------------------------------------------------------
__global__ void __launch_bounds__(32)
attn_kernel(const unsigned short* __restrict__ q,    // [H][N][DH], pre-scaled
            const unsigned short* __restrict__ kT,   // [H][DH][N]
            const unsigned short* __restrict__ vv,   // [H][N][DH]
            unsigned short* __restrict__ ao) {       // [N][C] bf16
  __shared__ __align__(16) unsigned short Psh[16 * 32];

  const int h    = blockIdx.y;
  const int n0   = blockIdx.x * 16;
  const int lane = threadIdx.x & 31;
  const int lm   = lane & 15;
  const int half = lane >> 4;
  const int kb   = half * 8;

  const int segs[6] = {0, 512, 1152, 1536, 1792, 2048};
  int s = 0;
  while (n0 >= segs[s + 1]) ++s;
  const int m_lo = segs[s], m_hi = segs[s + 1];

  // Q tile 16x64 as two A operands (d 0..31, 32..63), kept in registers
  ABuf qa0, qa1;
  {
    const unsigned short* qr = q + ((size_t)h * N_TOK + n0 + lm) * DH;
    qa0.f4[0] = *(const v4f*)(qr + kb);      qa0.f4[1] = *(const v4f*)(qr + kb + 16);
    qa1.f4[0] = *(const v4f*)(qr + 32 + kb); qa1.f4[1] = *(const v4f*)(qr + 32 + kb + 16);
  }

  v8f o0 = {}, o1 = {}, o2 = {}, o3 = {};
  float mi[8], li[8];
#pragma unroll
  for (int r = 0; r < 8; ++r) { mi[r] = -1.0e30f; li[r] = 0.0f; }

  for (int m0 = m_lo; m0 < m_hi; m0 += 32) {
    // S block 16x32 = two 16x16 WMMA accums, K reduced over Dh=64 in 2 chunks
    v8f s0 = {}, s1 = {};
#pragma unroll
    for (int dc = 0; dc < 2; ++dc) {
      const unsigned short* kr =
          kT + ((size_t)h * DH + dc * 32 + kb) * N_TOK + m0 + lm;   // B col = m, elements = d
      ABuf b0, b1;
      for (int j = 0; j < 16; ++j) {
        const size_t o = (size_t)((j & 7) + (j >> 3) * 16) * N_TOK;
        b0.v[j] = __builtin_bit_cast(__bf16, kr[o]);
        b1.v[j] = __builtin_bit_cast(__bf16, kr[o + 16]);
      }
      v16bf aq = dc ? qa1.v : qa0.v;
      s0 = wmma_bf16(aq, b0.v, s0);
      s1 = wmma_bf16(aq, b1.v, s1);
    }

    // Online softmax over these 32 columns
    float p0[8], p1[8];
#pragma unroll
    for (int r = 0; r < 8; ++r) {
      float a0 = s0[r], a1 = s1[r];
      float bm = fmaxf(a0, a1);
      for (int d = 8; d >= 1; d >>= 1) bm = fmaxf(bm, __shfl_xor(bm, d, 16));
      float mnew  = fmaxf(mi[r], bm);
      float alpha = __expf(mi[r] - mnew);
      float e0 = __expf(a0 - mnew);
      float e1 = __expf(a1 - mnew);
      float rs = e0 + e1;
      for (int d = 8; d >= 1; d >>= 1) rs += __shfl_xor(rs, d, 16);
      li[r] = li[r] * alpha + rs;
      mi[r] = mnew;
      o0[r] *= alpha; o1[r] *= alpha; o2[r] *= alpha; o3[r] *= alpha;
      p0[r] = e0; p1[r] = e1;
    }

    // P: C-layout -> A-operand layout via LDS bounce (row-major 16x32 bf16)
#pragma unroll
    for (int r = 0; r < 8; ++r) {
      Psh[(r + half * 8) * 32 + lm]      = f2bf(p0[r]);
      Psh[(r + half * 8) * 32 + 16 + lm] = f2bf(p1[r]);
    }
    __syncthreads();   // single-wave block: forces ds-cnt wait ordering
    ABuf pa;
    pa.f4[0] = *(const v4f*)&Psh[lm * 32 + kb];
    pa.f4[1] = *(const v4f*)&Psh[lm * 32 + kb + 16];
    __syncthreads();

    // O += P * V : B col = d (lane%16), halves = m rows {kb..kb+7, kb+16..kb+23}
    const unsigned short* vr = vv + ((size_t)h * N_TOK + m0 + kb) * DH + lm;
    ABuf bv;
    for (int j = 0; j < 16; ++j) bv.v[j] = __builtin_bit_cast(__bf16, vr[((j & 7) + (j >> 3) * 16) * DH +  0]);
    o0 = wmma_bf16(pa.v, bv.v, o0);
    for (int j = 0; j < 16; ++j) bv.v[j] = __builtin_bit_cast(__bf16, vr[((j & 7) + (j >> 3) * 16) * DH + 16]);
    o1 = wmma_bf16(pa.v, bv.v, o1);
    for (int j = 0; j < 16; ++j) bv.v[j] = __builtin_bit_cast(__bf16, vr[((j & 7) + (j >> 3) * 16) * DH + 32]);
    o2 = wmma_bf16(pa.v, bv.v, o2);
    for (int j = 0; j < 16; ++j) bv.v[j] = __builtin_bit_cast(__bf16, vr[((j & 7) + (j >> 3) * 16) * DH + 48]);
    o3 = wmma_bf16(pa.v, bv.v, o3);
  }

  // Normalize and store bf16 [N][C] for the proj GEMM
#pragma unroll
  for (int r = 0; r < 8; ++r) {
    const int n = n0 + r + half * 8;
    const float inv = 1.0f / li[r];
    unsigned short* orow = ao + (size_t)n * CDIM + h * DH + lm;
    orow[0]  = f2bf(o0[r] * inv);
    orow[16] = f2bf(o1[r] * inv);
    orow[32] = f2bf(o2[r] * inv);
    orow[48] = f2bf(o3[r] * inv);
  }
}

// ---------------------------------------------------------------------------
extern "C" void kernel_launch(void* const* d_in, const int* in_sizes, int n_in,
                              void* d_out, int out_size, void* d_ws, size_t ws_size,
                              hipStream_t stream) {
  const float* x      = (const float*)d_in[0];
  // d_in[1] = attn_bias: block-diagonal structure is static (SEG_LENS in the
  // reference), handled via segment loop bounds -> not read.
  const float* w_qkv  = (const float*)d_in[2];
  const float* w_proj = (const float*)d_in[3];
  const float* b_proj = (const float*)d_in[4];
  float* out = (float*)d_out;

  // Workspace carve-up (28 MiB total)
  char* ws = (char*)d_ws;
  unsigned short* xb  = (unsigned short*)ws; ws += (size_t)N_TOK * CDIM * 2;      // 4 MiB
  unsigned short* wqb = (unsigned short*)ws; ws += (size_t)CDIM * 3 * CDIM * 2;   // 6 MiB
  unsigned short* wpb = (unsigned short*)ws; ws += (size_t)CDIM * CDIM * 2;       // 2 MiB
  unsigned short* qb  = (unsigned short*)ws; ws += (size_t)HEADS * N_TOK * DH * 2;// 4 MiB
  unsigned short* kTb = (unsigned short*)ws; ws += (size_t)HEADS * DH * N_TOK * 2;// 4 MiB
  unsigned short* vb  = (unsigned short*)ws; ws += (size_t)HEADS * N_TOK * DH * 2;// 4 MiB
  unsigned short* aob = (unsigned short*)ws; ws += (size_t)N_TOK * CDIM * 2;      // 4 MiB

  int n4;
  n4 = (N_TOK * CDIM) / 4;
  cvt_f32_bf16<<<(n4 + 255) / 256, 256, 0, stream>>>(x, xb, n4);
  n4 = (CDIM * 3 * CDIM) / 4;
  cvt_f32_bf16<<<(n4 + 255) / 256, 256, 0, stream>>>(w_qkv, wqb, n4);
  n4 = (CDIM * CDIM) / 4;
  cvt_f32_bf16<<<(n4 + 255) / 256, 256, 0, stream>>>(w_proj, wpb, n4);

  // QKV: [2048x1024] x [1024x3072] -> scatter Q/K^T/V
  gemm_bf16<<<dim3(3 * CDIM / 64, N_TOK / 256), 256, 0, stream>>>(
      xb, wqb, CDIM, 3 * CDIM, 0, qb, kTb, vb, nullptr, nullptr);

  // Flash attention per (head, 16-row tile)
  attn_kernel<<<dim3(N_TOK / 16, HEADS), 32, 0, stream>>>(qb, kTb, vb, aob);

  // Projection: [2048x1024] x [1024x1024] + bias -> fp32 out
  gemm_bf16<<<dim3(CDIM / 64, N_TOK / 256), 256, 0, stream>>>(
      aob, wpb, CDIM, CDIM, 1, nullptr, nullptr, nullptr, out, b_proj);
}
